// ConservationConstrainedAttention_71631464563099
// MI455X (gfx1250) — matrix operand order repaired and checked by reading, not compiled, not run on hardware
//
#include <hip/hip_runtime.h>

typedef __bf16 bf16;
typedef __bf16 bf16x16 __attribute__((ext_vector_type(16)));
typedef float  f32x8   __attribute__((ext_vector_type(8)));
typedef int    i32x4   __attribute__((ext_vector_type(4)));
typedef unsigned int u32;

#define B_  4
#define S_  1024
#define D_  1024
#define H_  16
#define DH_ 64

// ---------------------------------------------------------------------------
// CDNA5 async global->LDS copy (ASYNCcnt-tracked), with safe fallbacks.
// Probe result: builtin exists with param0 = v4i in AS(1) (non-const).
// ---------------------------------------------------------------------------
#if defined(__has_builtin)
#if __has_builtin(__builtin_amdgcn_global_load_async_to_lds_b128)
#define HAVE_ASYNC_LDS 1
#endif
#endif

__device__ __forceinline__ void cp_b128(bf16* l, const bf16* g) {
#if defined(HAVE_ASYNC_LDS)
    __builtin_amdgcn_global_load_async_to_lds_b128(
        (__attribute__((address_space(1))) i32x4*)(g),
        (__attribute__((address_space(3))) i32x4*)(l), 0, 0);
#else
    *(uint4*)l = *(const uint4*)g;
#endif
}

__device__ __forceinline__ void async_wait() {
#if defined(HAVE_ASYNC_LDS)
#if __has_builtin(__builtin_amdgcn_s_wait_asynccnt)
    __builtin_amdgcn_s_wait_asynccnt(0);
#else
    asm volatile("s_wait_asynccnt 0x0" ::: "memory");
#endif
#endif
}

__device__ __forceinline__ f32x8 zero8() {
    f32x8 z = {0.f,0.f,0.f,0.f,0.f,0.f,0.f,0.f};
    return z;
}

__device__ __forceinline__ f32x8 wmma_bf16(bf16x16 a, bf16x16 b, f32x8 c) {
    return __builtin_amdgcn_wmma_f32_16x16x32_bf16(false, a, false, b, (short)0, c, false, false);
}

union Frag { bf16x16 v; u32 u[8]; };

// ---------------------------------------------------------------------------
// f32 -> bf16 conversion
// ---------------------------------------------------------------------------
__global__ void to_bf16_k(const float* __restrict__ in, bf16* __restrict__ out, int n) {
    int i = blockIdx.x * 256 + threadIdx.x;
    if (i < n) out[i] = (bf16)in[i];
}

// ---------------------------------------------------------------------------
// 128x128x32 bf16 WMMA GEMM with double-buffered async LDS tiles.
//   C[m,n] = sum_k A[m,k] * W[n,k] + bias[n]
// mode 0: write f32 row-major [M,N] to Cf
// mode 1: write bf16 head-split [B,H,S,DH] to Cb; if Ct != null also write
//         head-transposed bf16 [B,H,DH,S] (for the PV matmul B-operand).
// ---------------------------------------------------------------------------
__global__ __launch_bounds__(256) void gemm_bf16_k(
    const bf16* __restrict__ A, const bf16* __restrict__ W,
    const float* __restrict__ bias, bf16* __restrict__ Cb,
    bf16* __restrict__ Ct, float* __restrict__ Cf,
    int M, int N, int K, int mode)
{
    __shared__ __attribute__((aligned(16))) bf16 As[2 * 128 * 32];
    __shared__ __attribute__((aligned(16))) bf16 Ws[2 * 128 * 32];

    const int bm = blockIdx.y * 128;
    const int bn = blockIdx.x * 128;
    const int t    = threadIdx.x;
    const int wave = t >> 5, lane = t & 31;
    const int hi   = lane >> 4, r = lane & 15;
    const int wy = wave >> 1, wx = wave & 1;

    f32x8 acc[2][4];
    for (int i = 0; i < 2; i++)
        for (int j = 0; j < 4; j++) acc[i][j] = zero8();

    const int lrow = t >> 1;
    const int lcol = (t & 1) * 16;

    auto issue = [&](int buf, int kb) {
        const bf16* ga = A + (size_t)(bm + lrow) * K + kb + lcol;
        bf16* la = As + buf * 4096 + lrow * 32 + lcol;
        cp_b128(la, ga); cp_b128(la + 8, ga + 8);
        const bf16* gw = W + (size_t)(bn + lrow) * K + kb + lcol;
        bf16* lw = Ws + buf * 4096 + lrow * 32 + lcol;
        cp_b128(lw, gw); cp_b128(lw + 8, gw + 8);
    };

    issue(0, 0);
    async_wait();
    __syncthreads();

    for (int kb = 0; kb < K; kb += 32) {
        const int buf = (kb >> 5) & 1;
        if (kb + 32 < K) issue(buf ^ 1, kb + 32);
        if (kb + 64 < K) {  // prefetch tile-after-next into L2
            __builtin_prefetch(A + (size_t)(bm + lrow) * K + kb + 64 + lcol, 0, 1);
            __builtin_prefetch(W + (size_t)(bn + lrow) * K + kb + 64 + lcol, 0, 1);
        }

        const bf16* Ab = As + buf * 4096;
        const bf16* Wb = Ws + buf * 4096;
        Frag af[2], wf[4];
        #pragma unroll
        for (int ti = 0; ti < 2; ti++) {
            const u32* Ap = (const u32*)(Ab + (wy * 32 + ti * 16 + r) * 32);
            #pragma unroll
            for (int i = 0; i < 8; i++) {
                int koff = (i < 4) ? (hi * 8 + 2 * i) : (16 + hi * 8 + 2 * (i - 4));
                af[ti].u[i] = Ap[koff >> 1];
            }
        }
        #pragma unroll
        for (int tj = 0; tj < 4; tj++) {
            const u32* Wp = (const u32*)(Wb + (wx * 64 + tj * 16 + r) * 32);
            #pragma unroll
            for (int i = 0; i < 8; i++) {
                int koff = (i < 4) ? (hi * 8 + 2 * i) : (16 + hi * 8 + 2 * (i - 4));
                wf[tj].u[i] = Wp[koff >> 1];
            }
        }
        #pragma unroll
        for (int ti = 0; ti < 2; ti++)
            #pragma unroll
            for (int tj = 0; tj < 4; tj++)
                acc[ti][tj] = wmma_bf16(af[ti].v, wf[tj].v, acc[ti][tj]);

        async_wait();
        __syncthreads();
    }

    for (int ti = 0; ti < 2; ti++)
        for (int tj = 0; tj < 4; tj++) {
            #pragma unroll
            for (int v = 0; v < 8; v++) {
                int row = bm + wy * 32 + ti * 16 + v + hi * 8;
                int col = bn + wx * 64 + tj * 16 + r;
                float val = acc[ti][tj][v] + bias[col];
                if (mode == 0) {
                    Cf[(size_t)row * N + col] = val;
                } else {
                    int b = row >> 10, s = row & 1023;
                    int h = col >> 6,  dh = col & 63;
                    Cb[(((size_t)(b * H_ + h)) * S_ + s) * DH_ + dh] = (bf16)val;
                    if (Ct)
                        Ct[(((size_t)(b * H_ + h)) * DH_ + dh) * S_ + s] = (bf16)val;
                }
            }
        }
}

// ---------------------------------------------------------------------------
// energy_values[b,h,s] = V[b,h,s,:] . We + be
// ---------------------------------------------------------------------------
__global__ void energy_k(const bf16* __restrict__ Vh, const float* __restrict__ We,
                         const float* __restrict__ be, float* __restrict__ e)
{
    int i = blockIdx.x * 256 + threadIdx.x;
    if (i >= B_ * H_ * S_) return;
    float s = be[0];
    const bf16* v = Vh + (size_t)i * DH_;
    for (int d = 0; d < DH_; d++) s += (float)v[d] * We[d];
    e[i] = s;
}

__global__ void init_totals_k(float* totals) {
    if (threadIdx.x < B_) totals[threadIdx.x] = 0.f;
}

// ---------------------------------------------------------------------------
// Attention pass 1 (flash): first-softmax stats m,l + attended-energy totals.
// Grid: (S/128, B*H), 256 threads. Double-buffered async K tiles.
// ---------------------------------------------------------------------------
__global__ __launch_bounds__(256) void attn_pass1_k(
    const bf16* __restrict__ Qh, const bf16* __restrict__ Kh,
    const float* __restrict__ coords, const float* __restrict__ evals_g,
    float* __restrict__ mbuf, float* __restrict__ lbuf, float* __restrict__ totals)
{
    __shared__ __attribute__((aligned(16))) bf16 Qs[128 * 64];
    __shared__ __attribute__((aligned(16))) bf16 Ks[2 * 64 * 64];
    __shared__ float qc[128 * 4];
    __shared__ float kc[2][64 * 4];
    __shared__ float ev[2][64];

    const int bh = blockIdx.y;
    const int b  = bh >> 4;
    const int q0 = blockIdx.x * 128;
    const int t = threadIdx.x;
    const int wave = t >> 5, lane = t & 31, hi = lane >> 4, r = lane & 15;
    const float scale = 0.125f;

    {   // Q tile 128x64 (one-shot, synchronous vector copy)
        const uint4* src = (const uint4*)(Qh + ((size_t)bh * S_ + q0) * DH_) + t * 4;
        uint4* dst = (uint4*)Qs + t * 4;
        dst[0] = src[0]; dst[1] = src[1]; dst[2] = src[2]; dst[3] = src[3];
    }
    for (int i = t; i < 128 * 4; i += 256)
        qc[i] = coords[((size_t)b * S_ + q0 + (i >> 2)) * 4 + (i & 3)];
    __syncthreads();

    Frag qf[2];
    {
        const u32* Qp = (const u32*)(Qs + (wave * 16 + r) * 64);
        #pragma unroll
        for (int ks = 0; ks < 2; ks++)
            #pragma unroll
            for (int i = 0; i < 8; i++) {
                int koff = ks * 32 + ((i < 4) ? (hi * 8 + 2 * i) : (16 + hi * 8 + 2 * (i - 4)));
                qf[ks].u[i] = Qp[koff >> 1];
            }
    }

    auto issueK = [&](int buf, int it) {
        const bf16* g = Kh + ((size_t)bh * S_ + it * 64) * DH_ + t * 16;
        bf16* l = Ks + buf * 4096 + t * 16;
        cp_b128(l, g); cp_b128(l + 8, g + 8);
    };
    auto loadMeta = [&](int buf, int it) {
        int k0 = it * 64;
        if (t < 64) ev[buf][t] = evals_g[(size_t)bh * S_ + k0 + t];
        kc[buf][t] = coords[((size_t)b * S_ + k0 + (t >> 2)) * 4 + (t & 3)];
    };

    float m[8], l[8], ea[8];
    #pragma unroll
    for (int v = 0; v < 8; v++) { m[v] = -1e30f; l[v] = 0.f; ea[v] = 0.f; }

    const int NT = S_ / 64;
    issueK(0, 0);
    loadMeta(0, 0);
    async_wait();
    __syncthreads();

    for (int it = 0; it < NT; ++it) {
        const int buf = it & 1;
        if (it + 1 < NT) { issueK(buf ^ 1, it + 1); loadMeta(buf ^ 1, it + 1); }

        const bf16* Kb = Ks + buf * 4096;
        f32x8 sc[4];
        #pragma unroll
        for (int tj = 0; tj < 4; tj++) sc[tj] = zero8();
        #pragma unroll
        for (int ks = 0; ks < 2; ks++) {
            Frag kf[4];
            #pragma unroll
            for (int tj = 0; tj < 4; tj++) {
                const u32* Kp = (const u32*)(Kb + (tj * 16 + r) * 64);
                #pragma unroll
                for (int i = 0; i < 8; i++) {
                    int koff = ks * 32 + ((i < 4) ? (hi * 8 + 2 * i) : (16 + hi * 8 + 2 * (i - 4)));
                    kf[tj].u[i] = Kp[koff >> 1];
                }
            }
            #pragma unroll
            for (int tj = 0; tj < 4; tj++)
                sc[tj] = wmma_bf16(qf[ks].v, kf[tj].v, sc[tj]);
        }

        #pragma unroll
        for (int v = 0; v < 8; v++) {
            int qloc = wave * 16 + v + hi * 8;
            float c0 = qc[qloc*4+0], c1 = qc[qloc*4+1], c2 = qc[qloc*4+2], c3 = qc[qloc*4+3];
            float sv[4];
            #pragma unroll
            for (int tj = 0; tj < 4; tj++) {
                int kloc = tj * 16 + r;
                float dt = c0 - kc[buf][kloc*4+0];
                float d1 = c1 - kc[buf][kloc*4+1];
                float d2 = c2 - kc[buf][kloc*4+2];
                float d3 = c3 - kc[buf][kloc*4+3];
                float dr2 = d1*d1 + d2*d2 + d3*d3;
                float interval = dt*dt - dr2;
                float phys = __logf(__expf(-fabsf(interval) * 0.1f) + 1e-8f);
                bool causal = ((fabsf(dt) < __fsqrt_rn(dr2)) && (dr2 > 1e-6f)) || (dt < 0.f);
                sv[tj] = causal ? -1e30f : sc[tj][v] * scale + phys;
            }
            float smax = fmaxf(fmaxf(sv[0], sv[1]), fmaxf(sv[2], sv[3]));
            for (int mm = 1; mm < 16; mm <<= 1) smax = fmaxf(smax, __shfl_xor(smax, mm, 32));
            float mnew = fmaxf(m[v], smax);
            float rs = __expf(m[v] - mnew);
            float psum = 0.f, esum = 0.f;
            #pragma unroll
            for (int tj = 0; tj < 4; tj++) {
                float p = __expf(sv[tj] - mnew);
                psum += p;
                esum += p * ev[buf][tj * 16 + r];
            }
            for (int mm = 1; mm < 16; mm <<= 1) {
                psum += __shfl_xor(psum, mm, 32);
                esum += __shfl_xor(esum, mm, 32);
            }
            l[v]  = l[v]  * rs + psum;
            ea[v] = ea[v] * rs + esum;
            m[v]  = mnew;
        }
        async_wait();
        __syncthreads();
    }

    if (r == 0) {
        float tot = 0.f;
        #pragma unroll
        for (int v = 0; v < 8; v++) {
            int q = q0 + wave * 16 + v + hi * 8;
            size_t idx = (size_t)bh * S_ + q;
            mbuf[idx] = m[v];
            lbuf[idx] = l[v];
            tot += ea[v] / (l[v] + 1e-20f);
        }
        atomicAdd(&totals[b], tot);
    }
}

// ---------------------------------------------------------------------------
// Correction per batch
// ---------------------------------------------------------------------------
__global__ void corr_k(const float* __restrict__ totals, const float* __restrict__ E0,
                       float* __restrict__ corr)
{
    __shared__ float viol[B_];
    int t = threadIdx.x;
    if (t < B_) viol[t] = fabsf(totals[t] - E0[t]);
    __syncthreads();
    if (t < B_) {
        float mean = 0.f;
        for (int i = 0; i < B_; i++) mean += viol[i];
        mean *= (1.f / B_);
        corr[t] = (mean > 0.1f) ? (E0[t] / (totals[t] + 1e-8f)) : 1.0f;
    }
}

// ---------------------------------------------------------------------------
// Attention pass 2: rebuild w, apply correction, second softmax, O = P.V.
// V comes pre-transposed from HBM ([B,H,DH,S]); Q LDS region reused for P.
// Double-buffered async K / Vt tiles.
// ---------------------------------------------------------------------------
__global__ __launch_bounds__(256) void attn_pass2_k(
    const bf16* __restrict__ Qh, const bf16* __restrict__ Kh, const bf16* __restrict__ VtG,
    const float* __restrict__ coords, const float* __restrict__ mbuf,
    const float* __restrict__ lbuf, const float* __restrict__ corr,
    bf16* __restrict__ attended)
{
    __shared__ __attribute__((aligned(16))) char smem[53248];
    bf16* Qs  = (bf16*)smem;                     // 16 KB, reused as Ps after qf gather
    bf16* Ps  = (bf16*)smem;
    bf16* Ks  = (bf16*)(smem + 16384);           // 2 x 8 KB
    bf16* Vts = (bf16*)(smem + 32768);           // 2 x 8 KB  [dh][key]
    float* kc = (float*)(smem + 49152);          // 2 x 1 KB
    float* qc = (float*)(smem + 51200);          // 2 KB

    const int bh = blockIdx.y;
    const int b  = bh >> 4;
    const int h  = bh & 15;
    const int q0 = blockIdx.x * 128;
    const int t = threadIdx.x;
    const int wave = t >> 5, lane = t & 31, hi = lane >> 4, r = lane & 15;
    const float scale = 0.125f;
    const float cf = corr[b];

    {
        const uint4* src = (const uint4*)(Qh + ((size_t)bh * S_ + q0) * DH_) + t * 4;
        uint4* dst = (uint4*)Qs + t * 4;
        dst[0] = src[0]; dst[1] = src[1]; dst[2] = src[2]; dst[3] = src[3];
    }
    for (int i = t; i < 128 * 4; i += 256)
        qc[i] = coords[((size_t)b * S_ + q0 + (i >> 2)) * 4 + (i & 3)];
    __syncthreads();

    Frag qf[2];
    {
        const u32* Qp = (const u32*)(Qs + (wave * 16 + r) * 64);
        #pragma unroll
        for (int ks = 0; ks < 2; ks++)
            #pragma unroll
            for (int i = 0; i < 8; i++) {
                int koff = ks * 32 + ((i < 4) ? (hi * 8 + 2 * i) : (16 + hi * 8 + 2 * (i - 4)));
                qf[ks].u[i] = Qp[koff >> 1];
            }
    }

    auto issueTiles = [&](int buf, int it) {
        const bf16* gk = Kh + ((size_t)bh * S_ + it * 64) * DH_ + t * 16;
        bf16* lk = Ks + buf * 4096 + t * 16;
        cp_b128(lk, gk); cp_b128(lk + 8, gk + 8);
        // Vt tile: rows = dh (64), cols = keys (64), contiguous along keys
        int vrow = t >> 2, vcol = (t & 3) * 16;
        const bf16* gv = VtG + ((size_t)bh * DH_ + vrow) * S_ + it * 64 + vcol;
        bf16* lv = Vts + buf * 4096 + vrow * 64 + vcol;
        cp_b128(lv, gv); cp_b128(lv + 8, gv + 8);
    };
    auto loadMeta = [&](int buf, int it) {
        kc[buf * 256 + t] = coords[((size_t)b * S_ + it * 64 + (t >> 2)) * 4 + (t & 3)];
    };

    float mv[8], lv_[8], m2[8], l2[8];
    #pragma unroll
    for (int v = 0; v < 8; v++) {
        int q = q0 + wave * 16 + v + hi * 8;
        size_t idx = (size_t)bh * S_ + q;
        mv[v]  = mbuf[idx];
        lv_[v] = lbuf[idx] + 1e-20f;
        m2[v] = -1e30f;
        l2[v] = 0.f;
    }
    f32x8 o[4];
    #pragma unroll
    for (int tj = 0; tj < 4; tj++) o[tj] = zero8();

    const int NT = S_ / 64;
    issueTiles(0, 0);
    loadMeta(0, 0);
    async_wait();
    __syncthreads();   // also fences Qs reads before Ps reuse

    for (int it = 0; it < NT; ++it) {
        const int buf = it & 1;
        if (it + 1 < NT) { issueTiles(buf ^ 1, it + 1); loadMeta(buf ^ 1, it + 1); }

        const bf16* Kb  = Ks  + buf * 4096;
        const bf16* Vb  = Vts + buf * 4096;
        const float* kcb = kc + buf * 256;

        f32x8 sc[4];
        #pragma unroll
        for (int tj = 0; tj < 4; tj++) sc[tj] = zero8();
        #pragma unroll
        for (int ks = 0; ks < 2; ks++) {
            Frag kf[4];
            #pragma unroll
            for (int tj = 0; tj < 4; tj++) {
                const u32* Kp = (const u32*)(Kb + (tj * 16 + r) * 64);
                #pragma unroll
                for (int i = 0; i < 8; i++) {
                    int koff = ks * 32 + ((i < 4) ? (hi * 8 + 2 * i) : (16 + hi * 8 + 2 * (i - 4)));
                    kf[tj].u[i] = Kp[koff >> 1];
                }
            }
            #pragma unroll
            for (int tj = 0; tj < 4; tj++)
                sc[tj] = wmma_bf16(qf[ks].v, kf[tj].v, sc[tj]);
        }

        bf16* Pw = Ps + wave * 1024;
        #pragma unroll
        for (int v = 0; v < 8; v++) {
            int qloc = wave * 16 + v + hi * 8;
            float c0 = qc[qloc*4+0], c1 = qc[qloc*4+1], c2 = qc[qloc*4+2], c3 = qc[qloc*4+3];
            float wv[4];
            #pragma unroll
            for (int tj = 0; tj < 4; tj++) {
                int kloc = tj * 16 + r;
                float dt = c0 - kcb[kloc*4+0];
                float d1 = c1 - kcb[kloc*4+1];
                float d2 = c2 - kcb[kloc*4+2];
                float d3 = c3 - kcb[kloc*4+3];
                float dr2 = d1*d1 + d2*d2 + d3*d3;
                float interval = dt*dt - dr2;
                float phys = __logf(__expf(-fabsf(interval) * 0.1f) + 1e-8f);
                bool causal = ((fabsf(dt) < __fsqrt_rn(dr2)) && (dr2 > 1e-6f)) || (dt < 0.f);
                float s = causal ? -1e30f : sc[tj][v] * scale + phys;
                wv[tj] = __expf(s - mv[v]) / lv_[v] * cf;
            }
            float wmax = fmaxf(fmaxf(wv[0], wv[1]), fmaxf(wv[2], wv[3]));
            for (int mm = 1; mm < 16; mm <<= 1) wmax = fmaxf(wmax, __shfl_xor(wmax, mm, 32));
            float m2new = fmaxf(m2[v], wmax);
            float rs = __expf(m2[v] - m2new);
            float psum = 0.f;
            int wrow = v + hi * 8;
            #pragma unroll
            for (int tj = 0; tj < 4; tj++) {
                float p = __expf(wv[tj] - m2new);
                psum += p;
                Pw[wrow * 64 + tj * 16 + r] = (bf16)p;
            }
            for (int mm = 1; mm < 16; mm <<= 1) psum += __shfl_xor(psum, mm, 32);
            l2[v] = l2[v] * rs + psum;
            m2[v] = m2new;
            #pragma unroll
            for (int tj = 0; tj < 4; tj++) o[tj][v] *= rs;
        }

        // O += P x V (per-wave P tile; in-wave DS ordering)
        #pragma unroll
        for (int ks = 0; ks < 2; ks++) {
            Frag pf;
            {
                const u32* Pp = (const u32*)(Pw + r * 64);
                #pragma unroll
                for (int i = 0; i < 8; i++) {
                    int koff = ks * 32 + ((i < 4) ? (hi * 8 + 2 * i) : (16 + hi * 8 + 2 * (i - 4)));
                    pf.u[i] = Pp[koff >> 1];
                }
            }
            Frag vf[4];
            #pragma unroll
            for (int tj = 0; tj < 4; tj++) {
                const u32* Vp = (const u32*)(Vb + (tj * 16 + r) * 64);
                #pragma unroll
                for (int i = 0; i < 8; i++) {
                    int koff = ks * 32 + ((i < 4) ? (hi * 8 + 2 * i) : (16 + hi * 8 + 2 * (i - 4)));
                    vf[tj].u[i] = Vp[koff >> 1];
                }
            }
            #pragma unroll
            for (int tj = 0; tj < 4; tj++)
                o[tj] = wmma_bf16(pf.v, vf[tj].v, o[tj]);
        }
        async_wait();
        __syncthreads();
    }

    #pragma unroll
    for (int tj = 0; tj < 4; tj++) {
        #pragma unroll
        for (int v = 0; v < 8; v++) {
            int q = q0 + wave * 16 + v + hi * 8;
            int dh = tj * 16 + r;
            float val = o[tj][v] / (l2[v] + 1e-20f);
            attended[((size_t)b * S_ + q) * D_ + h * DH_ + dh] = (bf16)val;
        }
    }
}

// ---------------------------------------------------------------------------
// Residual + LayerNorm
// ---------------------------------------------------------------------------
__global__ __launch_bounds__(256) void ln_k(
    const float* __restrict__ outp, const float* __restrict__ query,
    const float* __restrict__ gamma, const float* __restrict__ beta,
    float* __restrict__ y)
{
    __shared__ float red[2][8];
    int row = blockIdx.x;
    int t = threadIdx.x;
    const float* o = outp  + (size_t)row * D_;
    const float* q = query + (size_t)row * D_;
    float s = 0.f, ss = 0.f, xv[4];
    #pragma unroll
    for (int i = 0; i < 4; i++) {
        float x = o[t + i * 256] + q[t + i * 256];
        xv[i] = x; s += x; ss += x * x;
    }
    for (int mm = 1; mm < 32; mm <<= 1) {
        s  += __shfl_xor(s,  mm, 32);
        ss += __shfl_xor(ss, mm, 32);
    }
    int wave = t >> 5, lane = t & 31;
    if (lane == 0) { red[0][wave] = s; red[1][wave] = ss; }
    __syncthreads();
    if (t == 0) {
        float a = 0.f, b2 = 0.f;
        for (int w = 0; w < 8; w++) { a += red[0][w]; b2 += red[1][w]; }
        red[0][0] = a; red[1][0] = b2;
    }
    __syncthreads();
    float mean = red[0][0] * (1.f / D_);
    float var  = red[1][0] * (1.f / D_) - mean * mean;
    float rstd = __frsqrt_rn(var + 1e-5f);
    #pragma unroll
    for (int i = 0; i < 4; i++) {
        int c = t + i * 256;
        y[(size_t)row * D_ + c] = (xv[i] - mean) * rstd * gamma[c] + beta[c];
    }
}

// ---------------------------------------------------------------------------
extern "C" void kernel_launch(void* const* d_in, const int* in_sizes, int n_in,
                              void* d_out, int out_size, void* d_ws, size_t ws_size,
                              hipStream_t stream)
{
    const float* query  = (const float*)d_in[0];
    const float* key    = (const float*)d_in[1];
    const float* value  = (const float*)d_in[2];
    const float* coords = (const float*)d_in[3];
    const float* E0     = (const float*)d_in[4];
    const float* Wq     = (const float*)d_in[5];
    const float* bq     = (const float*)d_in[6];
    const float* Wk     = (const float*)d_in[7];
    const float* bk     = (const float*)d_in[8];
    const float* Wv     = (const float*)d_in[9];
    const float* bv     = (const float*)d_in[10];
    const float* Wo     = (const float*)d_in[11];
    const float* bo     = (const float*)d_in[12];
    const float* We     = (const float*)d_in[13];
    const float* be     = (const float*)d_in[14];
    const float* gamma  = (const float*)d_in[15];
    const float* beta   = (const float*)d_in[16];

    char* ws = (char*)d_ws;
    size_t off = 0;
    auto alloc = [&](size_t bytes) -> void* {
        off = (off + 255) & ~(size_t)255;
        void* p = ws + off;
        off += bytes;
        return p;
    };
    const size_t nBSD = (size_t)B_ * S_ * D_;
    const size_t nDD  = (size_t)D_ * D_;
    const size_t nBHS = (size_t)B_ * H_ * S_;

    bf16* qb  = (bf16*)alloc(nBSD * 2);
    bf16* kb  = (bf16*)alloc(nBSD * 2);
    bf16* vb  = (bf16*)alloc(nBSD * 2);
    bf16* wqb = (bf16*)alloc(nDD * 2);
    bf16* wkb = (bf16*)alloc(nDD * 2);
    bf16* wvb = (bf16*)alloc(nDD * 2);
    bf16* wob = (bf16*)alloc(nDD * 2);
    bf16* Qh  = (bf16*)alloc(nBSD * 2);
    bf16* Kh  = (bf16*)alloc(nBSD * 2);
    bf16* Vh  = (bf16*)alloc(nBSD * 2);
    bf16* VtG = (bf16*)alloc(nBSD * 2);
    float* ebuf   = (float*)alloc(nBHS * 4);
    float* mbuf   = (float*)alloc(nBHS * 4);
    float* lbuf   = (float*)alloc(nBHS * 4);
    float* totals = (float*)alloc(B_ * 4);
    float* corr   = (float*)alloc(B_ * 4);
    bf16*  att    = (bf16*)alloc(nBSD * 2);
    float* outf   = (float*)alloc(nBSD * 4);

    const int nb1 = (int)((nBSD + 255) / 256);
    const int nb2 = (int)((nDD + 255) / 256);
    to_bf16_k<<<nb1, 256, 0, stream>>>(query, qb, (int)nBSD);
    to_bf16_k<<<nb1, 256, 0, stream>>>(key,   kb, (int)nBSD);
    to_bf16_k<<<nb1, 256, 0, stream>>>(value, vb, (int)nBSD);
    to_bf16_k<<<nb2, 256, 0, stream>>>(Wq, wqb, (int)nDD);
    to_bf16_k<<<nb2, 256, 0, stream>>>(Wk, wkb, (int)nDD);
    to_bf16_k<<<nb2, 256, 0, stream>>>(Wv, wvb, (int)nDD);
    to_bf16_k<<<nb2, 256, 0, stream>>>(Wo, wob, (int)nDD);

    dim3 gg(D_ / 128, (B_ * S_) / 128);
    gemm_bf16_k<<<gg, 256, 0, stream>>>(qb, wqb, bq, Qh, nullptr, nullptr, B_*S_, D_, D_, 1);
    gemm_bf16_k<<<gg, 256, 0, stream>>>(kb, wkb, bk, Kh, nullptr, nullptr, B_*S_, D_, D_, 1);
    gemm_bf16_k<<<gg, 256, 0, stream>>>(vb, wvb, bv, Vh, VtG,    nullptr, B_*S_, D_, D_, 1);

    energy_k<<<(int)((nBHS + 255) / 256), 256, 0, stream>>>(Vh, We, be, ebuf);
    init_totals_k<<<1, 32, 0, stream>>>(totals);

    dim3 ag(S_ / 128, B_ * H_);
    attn_pass1_k<<<ag, 256, 0, stream>>>(Qh, Kh, coords, ebuf, mbuf, lbuf, totals);
    corr_k<<<1, 32, 0, stream>>>(totals, E0, corr);
    attn_pass2_k<<<ag, 256, 0, stream>>>(Qh, Kh, VtG, coords, mbuf, lbuf, corr, att);

    gemm_bf16_k<<<gg, 256, 0, stream>>>(att, wob, bo, nullptr, nullptr, outf, B_*S_, D_, D_, 0);
    ln_k<<<B_ * S_, 256, 0, stream>>>(outf, query, gamma, beta, (float*)d_out);
}
